// STU_2319282340437
// MI455X (gfx1250) — matrix operand order repaired and checked by reading, not compiled, not run on hardware
//
#include <hip/hip_runtime.h>
#include <hip/hip_bf16.h>

// ============================================================================
// STU forward, MI455X (gfx1250, wave32, WMMA).
//
// Math plan (only the last timestep is needed):
//   delta[t,o] = sum_k sum_{sigma<=t} v'_k[t-sigma] * G_k[sigma,o]   (spectral)
//              + sum_lag sum_i u[t-lag,i] * m_u[o,i,lag]             (AR, fused)
//   where G_k = U @ m_phi_k  (lambda_k^{1/4} folded into v'_k)
//   y_t = A1 y_{t-1} + A2 y_{t-2} + delta_t  -> LTI chunk scan:
//   32 chunks x 64 steps, combined with C^64 (companion matrix power).
//
// Roofline: conv-as-block-Toeplitz = ~103 GFLOP bf16 -> WMMA-bound, tens of us.
// Gt (bf16, 100MB) lives in the 192MB L2, so the O(L^2) block re-reads never
// touch HBM; HBM moves ~120MB total (5us at 23.3TB/s).
// Workspace requirement: ~120 MB in d_ws.
// ============================================================================

#define L_SEQ   2048
#define DMODEL  256
#define KF      24
#define RVROW   2176     // L + 128 zero pad: covers block-diagonal tiles
#define NB      4
#define NCHUNK  32
#define CLEN    64       // NCHUNK * CLEN == L_SEQ

typedef __attribute__((ext_vector_type(16))) __bf16 v16bf;
typedef __attribute__((ext_vector_type(8)))  float  v8f;
typedef __attribute__((ext_vector_type(8)))  unsigned int v8u;

union FragU { v8u u; v16bf v; };

__device__ __forceinline__ unsigned short f2bf(float f) {
  unsigned int u = __float_as_uint(f);
  u += 0x7FFFu + ((u >> 16) & 1u);          // round-to-nearest-even
  return (unsigned short)(u >> 16);
}

__device__ __forceinline__ v16bf frag_zero() {
  FragU f; v8u z = {0u,0u,0u,0u,0u,0u,0u,0u}; f.u = z; return f.v;
}

// One lane's 16 bf16: 8 at p[0..7] and 8 at p[off2..off2+7] (both 16B aligned).
// A-frag (16x32): lane<16 -> K {0..7,16..23}, lane>=16 -> K {8..15,24..31}:
//   call with p = row + kbase (kbase=0/8), off2 = 16.
// B-frag (32x16): same K split as A-frag (lane<16 -> K {0..7,16..23}, else {8..15,24..31}):
//   call with p = col + cbase (cbase=0/8), off2 = 16.
__device__ __forceinline__ v16bf load_frag2(const unsigned short* p, int off2) {
  uint4 a = *reinterpret_cast<const uint4*>(p);
  uint4 b = *reinterpret_cast<const uint4*>(p + off2);
  FragU f;
  v8u t = {a.x, a.y, a.z, a.w, b.x, b.y, b.z, b.w};
  f.u = t;
  return f.v;
}

// 2-byte-aligned variant for Toeplitz A fragments (per-lane shifted windows).
__device__ __forceinline__ v16bf load_frag_u16(const unsigned short* p) {
  FragU f;
#pragma unroll
  for (int r = 0; r < 4; ++r) {
    f.u[r]     = (unsigned)p[2*r]    | ((unsigned)p[2*r+1]    << 16);
    f.u[r + 4] = (unsigned)p[16+2*r] | ((unsigned)p[16+2*r+1] << 16);
  }
  return f.v;
}

__device__ __forceinline__ v8f wmma_bf16(v16bf a, v16bf b, v8f c) {
  return __builtin_amdgcn_wmma_f32_16x16x32_bf16(false, a, false, b,
                                                 (short)0, c, false, false);
}

// ---------------------------------------------------------------------------
// Prep kernels
// ---------------------------------------------------------------------------

// rv[k][x] = lambda_k^{1/4} * v_k[L-1-x] for x<L, 0 for x in [L, RVROW).
// Then Toeplitz A[t_loc, kk] = v'_k[t - sigma] = rv[k][(L-1) - t + sigma0 + kk],
// ascending-contiguous in kk per lane; causal masking falls out of the zero pad.
__global__ void prep_rv(const float* __restrict__ eig_vals,
                        const float* __restrict__ eig_vecs,
                        unsigned short* __restrict__ rv) {
  int idx = blockIdx.x * blockDim.x + threadIdx.x;
  if (idx >= KF * RVROW) return;
  int k = idx / RVROW, x = idx % RVROW;
  float val = 0.f;
  if (x < L_SEQ) {
    float lam = powf(fabsf(eig_vals[k]), 0.25f);
    val = lam * eig_vecs[(L_SEQ - 1 - x) * KF + k];
  }
  rv[idx] = f2bf(val);
}

__global__ void cvt_bf16(const float* __restrict__ src,
                         unsigned short* __restrict__ dst, int n) {
  int i = blockIdx.x * blockDim.x + threadIdx.x;
  if (i < n) dst[i] = f2bf(src[i]);
}

// Pt[k][o][i] = bf16(m_phi[(k*D + i)*D + o])  (contraction i contiguous)
__global__ void prep_phi(const float* __restrict__ m_phi,
                         unsigned short* __restrict__ pt) {
  int idx = blockIdx.x * blockDim.x + threadIdx.x;
  if (idx >= KF * DMODEL * DMODEL) return;
  int k = idx / (DMODEL * DMODEL);
  int o = (idx / DMODEL) % DMODEL;
  int i = idx % DMODEL;
  pt[idx] = f2bf(m_phi[(k * DMODEL + i) * DMODEL + o]);
}

// Mut[lag][o][i] = bf16(m_u[o][i][lag])
__global__ void prep_mut(const float* __restrict__ m_u,
                         unsigned short* __restrict__ mut) {
  int idx = blockIdx.x * blockDim.x + threadIdx.x;
  if (idx >= 3 * DMODEL * DMODEL) return;
  int lag = idx / (DMODEL * DMODEL);
  int o = (idx / DMODEL) % DMODEL;
  int i = idx % DMODEL;
  mut[idx] = f2bf(m_u[(o * DMODEL + i) * 3 + lag]);
}

// Companion matrix C = [[A1, A2],[I, 0]] (512x512 f32)
__global__ void prep_cmat(const float* __restrict__ m_y, float* __restrict__ C) {
  int idx = blockIdx.x * blockDim.x + threadIdx.x;
  if (idx >= 512 * 512) return;
  int r = idx / 512, c = idx % 512;
  float v;
  if (r < DMODEL) v = m_y[r * 512 + c];     // m_y flat (o, ky, i) == row o, col ky*256+i
  else            v = (c == r - DMODEL) ? 1.f : 0.f;
  C[idx] = v;
}

// out = in @ in  (512x512 f32). grid (512, 2), 256 threads.
__global__ void matsq512(const float* __restrict__ in, float* __restrict__ out) {
  int r = blockIdx.x;
  int c = blockIdx.y * 256 + threadIdx.x;
  float acc = 0.f;
  for (int j = 0; j < 512; ++j) acc += in[r * 512 + j] * in[j * 512 + c];
  out[r * 512 + c] = acc;
}

// ---------------------------------------------------------------------------
// G_k = U @ m_phi_k, stored TRANSPOSED+bf16: Gt[b][k][o][sigma]
// (C-fragment rows are sigma-contiguous -> one b128 store per lane; later
//  B-fragments read two contiguous b128 per lane.)
// grid (6144, NB), 256 threads (8 waves), 1 16x16 tile per wave, K-loop 8 WMMAs.
// ---------------------------------------------------------------------------
__global__ void gemm_g(const unsigned short* __restrict__ Ub,
                       const unsigned short* __restrict__ Pt,
                       unsigned short* __restrict__ Gt) {
  int lane = threadIdx.x & 31;
  int wave = threadIdx.x >> 5;
  int b = blockIdx.y;
  int tile = blockIdx.x * 8 + wave;              // < KF*128*16
  int k  = tile >> 11;
  int r  = tile & 2047;
  int st = r >> 4;                               // sigma tile (0..127)
  int ot = r & 15;                               // o tile (0..15)

  int m  = lane & 15;
  int kb = (lane < 16) ? 0 : 8;                  // A contraction base
  int bb = (lane < 16) ? 0 : 8;                  // B contraction base
  int o  = ot * 16 + (lane & 15);

  const unsigned short* arow = Ub + ((size_t)b * L_SEQ + st * 16 + m) * DMODEL;
  const unsigned short* brow = Pt + ((size_t)k * DMODEL + o) * DMODEL;

  v8f acc = {0.f,0.f,0.f,0.f,0.f,0.f,0.f,0.f};
#pragma unroll
  for (int i0 = 0; i0 < DMODEL; i0 += 32) {
    v16bf a = load_frag2(arow + i0 + kb, 16);
    v16bf bm = load_frag2(brow + i0 + bb, 16);
    acc = wmma_bf16(a, bm, acc);
  }

  // store bf16, sigma-contiguous: lane<16 covers sigma sb..sb+7, lane>=16 +8..+15
  int sb = st * 16 + ((lane < 16) ? 0 : 8);
  unsigned short* drow = Gt + (((size_t)b * KF + k) * DMODEL + o) * L_SEQ + sb;
  uint4 out;
  out.x = (unsigned)f2bf(acc[0]) | ((unsigned)f2bf(acc[1]) << 16);
  out.y = (unsigned)f2bf(acc[2]) | ((unsigned)f2bf(acc[3]) << 16);
  out.z = (unsigned)f2bf(acc[4]) | ((unsigned)f2bf(acc[5]) << 16);
  out.w = (unsigned)f2bf(acc[6]) | ((unsigned)f2bf(acc[7]) << 16);
  *reinterpret_cast<uint4*>(drow) = out;
}

// ---------------------------------------------------------------------------
// Main kernel: causal block-Toeplitz conv (sum over k) + fused AR term.
// grid (16 t-blocks, 2 o-halves, NB batches), 256 threads (8 waves).
// Wave w: t-strip [tb+16w, +16) x 128 output cols -> acc[8] (64 VGPRs).
// ---------------------------------------------------------------------------
__global__ void conv_main(const unsigned short* __restrict__ rv,
                          const unsigned short* __restrict__ Gt,
                          const unsigned short* __restrict__ Ub,
                          const unsigned short* __restrict__ Mut,
                          float* __restrict__ delta) {
  int lane = threadIdx.x & 31;
  int wave = threadIdx.x >> 5;
  int tb = blockIdx.x * 128;
  int ob = blockIdx.y * 128;
  int b  = blockIdx.z;
  int tw = tb + wave * 16;

  int m     = lane & 15;
  int t     = tw + m;
  int kb    = (lane < 16) ? 0 : 8;               // A contraction base
  int bb    = (lane < 16) ? 0 : 8;               // B contraction base
  int ncol  = lane & 15;

  const unsigned short* Gt_b = Gt + (size_t)b * KF * DMODEL * L_SEQ;
  const unsigned short* Ub_b = Ub + (size_t)b * L_SEQ * DMODEL;

  v8f acc[8];
#pragma unroll
  for (int n = 0; n < 8; ++n) acc[n] = (v8f){0.f,0.f,0.f,0.f,0.f,0.f,0.f,0.f};

  // -------- spectral part: sum_k (T_k @ G_k), causal via rv zero padding -----
  int smax = tb + 128;
  int rvbase = (L_SEQ - 1) - t + kb;             // per-lane, loop-invariant
  for (int s0 = 0; s0 < smax; s0 += 32) {
    for (int k = 0; k < KF; ++k) {
      const unsigned short* ap = rv + k * RVROW + rvbase + s0;
      v16bf a = load_frag_u16(ap);
      const unsigned short* gk = Gt_b + (size_t)k * DMODEL * L_SEQ;
#pragma unroll
      for (int n = 0; n < 8; ++n) {
        const unsigned short* bp = gk + (size_t)(ob + n * 16 + ncol) * L_SEQ + s0 + bb;
        __builtin_prefetch(bp + 32, 0, 1);       // next sigma block -> global_prefetch
        v16bf bm = load_frag2(bp, 16);
        acc[n] = wmma_bf16(a, bm, acc[n]);
      }
    }
  }

  // -------- fused AR term: delta += sum_lag U_shift(lag) @ Mut[lag]^T --------
  for (int lag = 0; lag < 3; ++lag) {
    int ta = t - lag;                            // per-lane row, guard t<lag
#pragma unroll
    for (int i0 = 0; i0 < DMODEL; i0 += 32) {
      v16bf a;
      if (ta >= 0) a = load_frag2(Ub_b + (size_t)ta * DMODEL + i0 + kb, 16);
      else         a = frag_zero();
#pragma unroll
      for (int n = 0; n < 8; ++n) {
        const unsigned short* bp =
            Mut + ((size_t)lag * DMODEL + ob + n * 16 + ncol) * DMODEL + i0 + bb;
        acc[n] = wmma_bf16(a, load_frag2(bp, 16), acc[n]);
      }
    }
  }

  // -------- store f32 delta[b][t][o] -----------------------------------------
  float* db = delta + (size_t)b * L_SEQ * DMODEL;
  int trow = tw + ((lane < 16) ? 0 : 8);
#pragma unroll
  for (int n = 0; n < 8; ++n) {
#pragma unroll
    for (int r = 0; r < 8; ++r)
      db[(size_t)(trow + r) * DMODEL + ob + n * 16 + ncol] = acc[n][r];
  }
}

// ---------------------------------------------------------------------------
// In-chunk recurrence: 128 independent scans (32 chunks x 4 batches), 64 steps.
// first 256 bf16 of m_y row (A1 half) pinned in 32 uint4 registers per lane; carry in LDS f32.
// ---------------------------------------------------------------------------
__global__ void chunk_scan(const unsigned short* __restrict__ Myb,
                           const float* __restrict__ delta,
                           float* __restrict__ Z) {
  __shared__ __align__(16) float carry[512];
  int o = threadIdx.x;                           // 256 threads
  int c = blockIdx.x;
  int b = blockIdx.y;

  uint4 w[32];
  const uint4* wp = reinterpret_cast<const uint4*>(Myb + (size_t)o * 512);
#pragma unroll
  for (int q = 0; q < 32; ++q) w[q] = wp[q];

  carry[o] = 0.f; carry[o + 256] = 0.f;
  __syncthreads();

  const float* db = delta + ((size_t)b * L_SEQ + c * CLEN) * DMODEL;
  for (int step = 0; step < CLEN; ++step) {
    float acc = db[(size_t)step * DMODEL + o];
    const float4* cf = reinterpret_cast<const float4*>(carry);
#pragma unroll
    for (int q = 0; q < 32; ++q) {               // j = 8q .. 8q+7
      float4 c0 = cf[q * 2], c1 = cf[q * 2 + 1];
      unsigned x;
      x = w[q].x; acc += __uint_as_float(x << 16) * c0.x + __uint_as_float(x & 0xffff0000u) * c0.y;
      x = w[q].y; acc += __uint_as_float(x << 16) * c0.z + __uint_as_float(x & 0xffff0000u) * c0.w;
      x = w[q].z; acc += __uint_as_float(x << 16) * c1.x + __uint_as_float(x & 0xffff0000u) * c1.y;
      x = w[q].w; acc += __uint_as_float(x << 16) * c1.z + __uint_as_float(x & 0xffff0000u) * c1.w;
    }
    __syncthreads();                             // all reads of carry done
    float prev0 = carry[o];
    carry[o + 256] = prev0;
    carry[o] = acc;
    __syncthreads();
  }
  Z[((size_t)b * NCHUNK + c) * 512 + o]       = carry[o];
  Z[((size_t)b * NCHUNK + c) * 512 + 256 + o] = carry[o + 256];
}

// Combine: s = z_0; for c=1..NC-1: s = C64 @ s + z_c.  grid(NB), 512 threads.
__global__ void combine(const float* __restrict__ C64,
                        const float* __restrict__ Z,
                        float* __restrict__ out) {
  __shared__ float s[512];
  int tid = threadIdx.x;
  int b = blockIdx.x;
  s[tid] = Z[(size_t)b * NCHUNK * 512 + tid];
  __syncthreads();
  for (int c = 1; c < NCHUNK; ++c) {
    const float* row = C64 + (size_t)tid * 512;
    float acc = Z[((size_t)b * NCHUNK + c) * 512 + tid];
    for (int j = 0; j < 512; ++j) acc += row[j] * s[j];
    __syncthreads();
    s[tid] = acc;
    __syncthreads();
  }
  if (tid < DMODEL) out[b * DMODEL + tid] = s[tid];
}

// ---------------------------------------------------------------------------
extern "C" void kernel_launch(void* const* d_in, const int* in_sizes, int n_in,
                              void* d_out, int out_size, void* d_ws, size_t ws_size,
                              hipStream_t stream) {
  const float* inputs  = (const float*)d_in[0];   // (B, L, D)
  const float* eigvals = (const float*)d_in[1];   // (K)
  const float* eigvecs = (const float*)d_in[2];   // (L, K)
  const float* m_u     = (const float*)d_in[3];   // (D, D, 3)
  const float* m_phi   = (const float*)d_in[4];   // (K*D, D)
  const float* m_y     = (const float*)d_in[5];   // (D, 2, D)
  float* out = (float*)d_out;

  char* ws = (char*)d_ws;
  size_t off = 0;
  auto alloc = [&](size_t bytes) -> void* {
    void* p = ws + off;
    off = (off + bytes + 1023) & ~(size_t)1023;
    return p;
  };

  unsigned short* rv   = (unsigned short*)alloc((size_t)KF * RVROW * 2);
  unsigned short* Ub   = (unsigned short*)alloc((size_t)NB * L_SEQ * DMODEL * 2);
  unsigned short* Pt   = (unsigned short*)alloc((size_t)KF * DMODEL * DMODEL * 2);
  unsigned short* Mut  = (unsigned short*)alloc((size_t)3 * DMODEL * DMODEL * 2);
  unsigned short* Myb  = (unsigned short*)alloc((size_t)DMODEL * 512 * 2);
  float*          CA   = (float*)alloc((size_t)512 * 512 * 4);
  float*          CB   = (float*)alloc((size_t)512 * 512 * 4);
  unsigned short* Gt   = (unsigned short*)alloc((size_t)NB * KF * DMODEL * L_SEQ * 2);
  float*          delta= (float*)alloc((size_t)NB * L_SEQ * DMODEL * 4);
  float*          Z    = (float*)alloc((size_t)NB * NCHUNK * 512 * 4);
  (void)ws_size; (void)in_sizes; (void)n_in; (void)out_size;   // ~120 MB used

  // ---- prep ----
  prep_rv  <<<(KF * RVROW + 255) / 256, 256, 0, stream>>>(eigvals, eigvecs, rv);
  cvt_bf16 <<<(NB * L_SEQ * DMODEL + 255) / 256, 256, 0, stream>>>(inputs, Ub, NB * L_SEQ * DMODEL);
  prep_phi <<<(KF * DMODEL * DMODEL + 255) / 256, 256, 0, stream>>>(m_phi, Pt);
  prep_mut <<<(3 * DMODEL * DMODEL + 255) / 256, 256, 0, stream>>>(m_u, Mut);
  cvt_bf16 <<<(DMODEL * 512 + 255) / 256, 256, 0, stream>>>(m_y, Myb, DMODEL * 512);
  prep_cmat<<<(512 * 512 + 255) / 256, 256, 0, stream>>>(m_y, CA);

  // ---- C^64 via 6 squarings: CA -> CB -> CA -> ... -> CA ----
  for (int i = 0; i < 6; ++i) {
    const float* in = (i & 1) ? CB : CA;
    float*      o2 = (i & 1) ? CA : CB;
    matsq512<<<dim3(512, 2), 256, 0, stream>>>(in, o2);
  }

  // ---- G = U @ m_phi_k (bf16 WMMA), then the big Toeplitz WMMA conv ----
  gemm_g   <<<dim3(6144, NB), 256, 0, stream>>>(Ub, Pt, Gt);
  conv_main<<<dim3(16, 2, NB), 256, 0, stream>>>(rv, Gt, Ub, Mut, delta);

  // ---- chunked LTI scan + combine ----
  chunk_scan<<<dim3(NCHUNK, NB), 256, 0, stream>>>(Myb, delta, Z);
  combine   <<<NB, 512, 0, stream>>>(CA, Z, out);
}
